// ClassicalSelfAttention_65481071400281
// MI455X (gfx1250) — matrix operand order repaired and checked by reading, not compiled, not run on hardware
//
#include <hip/hip_runtime.h>
#include <hip/hip_bf16.h>
#include <math.h>

// ---------------------------------------------------------------------------
// CDNA5 (gfx1250) wave32 WMMA types
// ---------------------------------------------------------------------------
typedef __attribute__((ext_vector_type(16))) __bf16 bf16x16;
typedef __attribute__((ext_vector_type(8)))  float  f32x8;

union FragAB {
    uint4   u[2];   // two 16B chunks per lane (16 bf16)
    bf16x16 v;
};

__device__ __forceinline__ f32x8 zero8() {
    f32x8 z;
    for (int i = 0; i < 8; ++i) z[i] = 0.0f;
    return z;
}

__device__ __forceinline__ f32x8 wmma_bf16(bf16x16 a, bf16x16 b, f32x8 c) {
    // D = A(16x32 bf16) * B(32x16 bf16) + C(16x16 f32)
    return __builtin_amdgcn_wmma_f32_16x16x32_bf16(false, a, false, b,
                                                   (short)0, c, false, false);
}

// round-to-nearest-even f32 -> bf16
__device__ __forceinline__ unsigned short f2bf(float f) {
    unsigned u = __float_as_uint(f);
    u += 0x7FFFu + ((u >> 16) & 1u);
    return (unsigned short)(u >> 16);
}

// ---------------------------------------------------------------------------
// CDNA5 async global->LDS copy (ASYNCcnt-tracked, bypasses VGPRs).
// vdst = LDS byte address (flat LDS address truncates to the DS offset),
// vaddr = 64-bit global address, saddr = off.  (ISA 15.18.3 opcode 98)
// ---------------------------------------------------------------------------
__device__ __forceinline__ unsigned lds_off(const void* p) {
    return (unsigned)(size_t)p;   // flat LDS aperture: addr[31:0] is DS address
}

__device__ __forceinline__ void async_copy_b128(const void* gsrc, unsigned ldsaddr) {
    asm volatile("global_load_async_to_lds_b128 %0, %1, off"
                 :
                 : "v"(ldsaddr), "v"(gsrc)
                 : "memory");
}

__device__ __forceinline__ void wait_async_all() {
    asm volatile("s_wait_asynccnt 0x0" ::: "memory");
}

// ---------------------------------------------------------------------------
// Kernel 0: f32 -> bf16 conversion (vectorized, 4 elems/thread)
// ---------------------------------------------------------------------------
__global__ void cvt_f32_bf16(const float* __restrict__ in,
                             unsigned short* __restrict__ out, int n4) {
    int i = blockIdx.x * blockDim.x + threadIdx.x;
    if (i < n4) {
        float4 f = ((const float4*)in)[i];
        uint2 o;
        o.x = (unsigned)f2bf(f.x) | ((unsigned)f2bf(f.y) << 16);
        o.y = (unsigned)f2bf(f.z) | ((unsigned)f2bf(f.w) << 16);
        ((uint2*)out)[i] = o;
    }
}

// ---------------------------------------------------------------------------
// Kernel 1: C(bf16)[8192,1024] = A(bf16)[8192,1024] @ B(bf16)[1024,1024]
// blockIdx.z selects (rotation->Q) or (entangle->K).
// Block tile 128x128, 256 threads = 8 waves, each wave 32x64 (2x4 WMMA tiles).
// A tile staged with async LDS copies; B staged transposed via VALU packing.
// ---------------------------------------------------------------------------
__global__ __launch_bounds__(256) void gemm_bf16_qk(
    const unsigned short* __restrict__ A,
    const unsigned short* __restrict__ B0,
    const unsigned short* __restrict__ B1,
    unsigned short* __restrict__ C0,
    unsigned short* __restrict__ C1)
{
    const unsigned short* B = blockIdx.z ? B1 : B0;
    unsigned short*       C = blockIdx.z ? C1 : C0;
    const int m0 = blockIdx.y * 128;
    const int n0 = blockIdx.x * 128;
    const int t    = threadIdx.x;
    const int lane = t & 31;
    const int w    = t >> 5;
    const int lm   = lane & 15;
    const int wm   = w & 3;        // 4 M-chunks of 32 rows
    const int wn   = w >> 2;       // 2 N-chunks of 64 cols
    const int klo  = (lane < 16) ? 0 : 16;   // A-frag k base (contiguous k = 16h + s)
    const int khi  = (lane < 16) ? 0 : 16;   // B-frag k base (contiguous k = 16h + s, same as A)

    __shared__ __align__(16) unsigned short As[128 * 72];  // [row][k], stride 72
    __shared__ __align__(16) unsigned short BT[128 * 72];  // [n][k],  stride 72

    f32x8 acc[2][4];
    for (int mt = 0; mt < 2; ++mt)
        for (int nt = 0; nt < 4; ++nt) acc[mt][nt] = zero8();

    for (int kk = 0; kk < 1024; kk += 64) {
        // stage A tile 128x64: 1024 x 16B async copies (no VGPR round-trip)
        for (int i = 0; i < 4; ++i) {
            int c  = t + 256 * i;
            int r  = c >> 3;
            int c8 = (c & 7) * 8;
            async_copy_b128(&A[(size_t)(m0 + r) * 1024 + kk + c8],
                            lds_off(&As[r * 72 + c8]));
        }
        // stage B transposed: BT[n][k] = B[k][n]; pack (k,k+1) pairs as u32
        for (int i = 0; i < 4; ++i) {
            int u   = t + 256 * i;
            int kp  = u >> 5;          // 0..31 -> k = 2*kp
            int ncl = u & 31;          // col group of 4
            const unsigned short* bp =
                &B[(size_t)(kk + 2 * kp) * 1024 + n0 + 4 * ncl];
            uint2 r0 = *(const uint2*)bp;
            uint2 r1 = *(const uint2*)(bp + 1024);
            unsigned e0[4] = { r0.x & 0xFFFFu, r0.x >> 16, r0.y & 0xFFFFu, r0.y >> 16 };
            unsigned e1[4] = { r1.x & 0xFFFFu, r1.x >> 16, r1.y & 0xFFFFu, r1.y >> 16 };
            for (int c = 0; c < 4; ++c)
                *(unsigned*)&BT[(4 * ncl + c) * 72 + 2 * kp] = e0[c] | (e1[c] << 16);
        }
        wait_async_all();
        __syncthreads();

        for (int ks = 0; ks < 64; ks += 32) {
            FragAB fa[2], fb[4];
            for (int mt = 0; mt < 2; ++mt) {
                int row = wm * 32 + mt * 16 + lm;
                fa[mt].u[0] = *(const uint4*)&As[row * 72 + ks + klo];
                fa[mt].u[1] = *(const uint4*)&As[row * 72 + ks + klo + 8];
            }
            for (int nt = 0; nt < 4; ++nt) {
                int n = wn * 64 + nt * 16 + lm;
                fb[nt].u[0] = *(const uint4*)&BT[n * 72 + ks + khi];
                fb[nt].u[1] = *(const uint4*)&BT[n * 72 + ks + khi + 8];
            }
            for (int mt = 0; mt < 2; ++mt)
                for (int nt = 0; nt < 4; ++nt)
                    acc[mt][nt] = wmma_bf16(fa[mt].v, fb[nt].v, acc[mt][nt]);
        }
        __syncthreads();
    }

    // epilogue: f32 acc -> bf16 store (ISA 16x16 f32 C layout)
    const int hi8 = (lane < 16) ? 0 : 8;
    for (int mt = 0; mt < 2; ++mt)
        for (int nt = 0; nt < 4; ++nt)
            for (int r = 0; r < 8; ++r) {
                int row = m0 + wm * 32 + mt * 16 + r + hi8;
                int col = n0 + wn * 64 + nt * 16 + lm;
                C[(size_t)row * 1024 + col] = f2bf(acc[mt][nt][r]);
            }
}

// ---------------------------------------------------------------------------
// Kernel 2/3: streaming attention. Each block owns 32 query rows (Q resident
// in LDS via async copies), streams 128 key-blocks of 64 rows (K chunks via
// async copies + prefetch of the next block). STATS pass computes per-row
// running max / sum-exp; OUT pass normalizes with precomputed stats and
// accumulates O += P @ V via WMMA (V staged transposed in LDS, quarter-D at
// a time; each wave owns a 32-column output slice per quarter).
// ---------------------------------------------------------------------------
template <bool WRITE_OUT>
__global__ __launch_bounds__(256) void attn_kernel(
    const unsigned short* __restrict__ Qb,
    const unsigned short* __restrict__ Kb,
    const unsigned short* __restrict__ Vb,
    float* __restrict__ gM, float* __restrict__ gS,
    float* __restrict__ out, float scale)
{
    const int m0   = blockIdx.x * 32;
    const int t    = threadIdx.x;
    const int lane = t & 31;
    const int w    = t >> 5;
    const int lm   = lane & 15;
    const int mi   = w >> 2;          // 0..1 : 16-row S tile
    const int nq   = w & 3;           // 0..3 : 16-col S tile
    const int hi8  = (lane < 16) ? 0 : 8;
    const int klo  = (lane < 16) ? 0 : 16;
    const int khi  = (lane < 16) ? 0 : 16;

    __shared__ __align__(16) unsigned short Qs[32 * 1032];  // Q block, padded
    __shared__ __align__(16) unsigned short Ks[64 * 72];    // K chunk [key][k]
    __shared__ __align__(16) float          Sbuf[32 * 64];  // logits
    __shared__ __align__(16) unsigned short Pbuf[32 * 72];  // probs bf16
    __shared__ __align__(16) unsigned short VT[256 * 72];   // V^T quarter [n][k]
    __shared__ float red[32 * 9];                           // row reductions

    // stage Q block (32 x 1024 bf16) once, asynchronously
    for (int i = 0; i < 16; ++i) {
        int c  = t + 256 * i;
        int r  = c >> 7;
        int c8 = (c & 127) * 8;
        async_copy_b128(&Qb[(size_t)(m0 + r) * 1024 + c8],
                        lds_off(&Qs[r * 1032 + c8]));
    }

    const int ri  = t >> 3;        // row 0..31 for softmax phase
    const int c8i = t & 7;         // 8 threads per row
    const int c0  = c8i * 8;

    float m_run = -INFINITY, s_run = 0.0f;
    float myM = 0.0f, myInvS = 1.0f;
    if constexpr (WRITE_OUT) {
        myM    = gM[m0 + ri];
        myInvS = 1.0f / gS[m0 + ri];
    }

    f32x8 oacc[4][2][2];
    if constexpr (WRITE_OUT) {
        for (int q = 0; q < 4; ++q)
            for (int mt = 0; mt < 2; ++mt)
                for (int nt = 0; nt < 2; ++nt) oacc[q][mt][nt] = zero8();
    }

    const int krow = t >> 2;              // K-chunk staging row 0..63
    const int kcb  = (t & 3) * 16;        // K-chunk staging col 0/16/32/48

    for (int j = 0; j < 128; ++j) {
        // ----- S = Q @ K^T for key rows [j*64, j*64+64) -----
        f32x8 sacc = zero8();
        for (int kk = 0; kk < 1024; kk += 64) {
            {   // stage K chunk 64x64 via async LDS copies
                const unsigned short* kp =
                    &Kb[(size_t)(j * 64 + krow) * 1024 + kk + kcb];
                async_copy_b128(kp,     lds_off(&Ks[krow * 72 + kcb]));
                async_copy_b128(kp + 8, lds_off(&Ks[krow * 72 + kcb + 8]));
                if (j + 1 < 128)  // pull next block's K toward the WGP caches
                    __builtin_prefetch(
                        &Kb[(size_t)((j + 1) * 64 + krow) * 1024 + kk + kcb], 0, 3);
            }
            wait_async_all();
            __syncthreads();
            for (int ks = 0; ks < 64; ks += 32) {
                FragAB fa, fb;
                int arow = mi * 16 + lm;
                fa.u[0] = *(const uint4*)&Qs[arow * 1032 + kk + ks + klo];
                fa.u[1] = *(const uint4*)&Qs[arow * 1032 + kk + ks + klo + 8];
                int n = nq * 16 + lm;
                fb.u[0] = *(const uint4*)&Ks[n * 72 + ks + khi];
                fb.u[1] = *(const uint4*)&Ks[n * 72 + ks + khi + 8];
                sacc = wmma_bf16(fa.v, fb.v, sacc);
            }
            __syncthreads();
        }
        for (int r = 0; r < 8; ++r)
            Sbuf[(mi * 16 + r + hi8) * 64 + nq * 16 + lm] = sacc[r] * scale;
        __syncthreads();

        if constexpr (!WRITE_OUT) {
            // ----- online row stats -----
            float lmax = -INFINITY;
            for (int c = 0; c < 8; ++c)
                lmax = fmaxf(lmax, Sbuf[ri * 64 + c0 + c]);
            red[ri * 9 + 1 + c8i] = lmax;
            __syncthreads();
            if (c8i == 0) {
                float bm = red[ri * 9 + 1];
                for (int c = 2; c <= 8; ++c) bm = fmaxf(bm, red[ri * 9 + c]);
                red[ri * 9] = bm;
            }
            __syncthreads();
            float bmax = red[ri * 9];
            float lsum = 0.0f;
            for (int c = 0; c < 8; ++c)
                lsum += __expf(Sbuf[ri * 64 + c0 + c] - bmax);
            red[ri * 9 + 1 + c8i] = lsum;
            __syncthreads();
            if (c8i == 0) {
                float bs = 0.0f;
                for (int c = 1; c <= 8; ++c) bs += red[ri * 9 + c];
                float nm = fmaxf(m_run, bmax);
                s_run = s_run * __expf(m_run - nm) + bs * __expf(bmax - nm);
                m_run = nm;
            }
            __syncthreads();
        } else {
            // ----- P = exp(S - m)/sum in bf16 -----
            for (int c = 0; c < 8; c += 2) {
                float p0 = __expf(Sbuf[ri * 64 + c0 + c]     - myM) * myInvS;
                float p1 = __expf(Sbuf[ri * 64 + c0 + c + 1] - myM) * myInvS;
                *(unsigned*)&Pbuf[ri * 72 + c0 + c] =
                    (unsigned)f2bf(p0) | ((unsigned)f2bf(p1) << 16);
            }
            // ----- O += P @ V, quarter of D at a time -----
            for (int q = 0; q < 4; ++q) {
                // stage VT[n][k] = V[j*64+k][q*256+n], (k,k+1) packed as u32
                for (int i = 0; i < 8; ++i) {
                    int u   = t + 256 * i;
                    int kp  = u >> 6;        // 0..31 -> key 2*kp
                    int ncl = u & 63;        // col group of 4
                    const unsigned short* vp =
                        &Vb[(size_t)(j * 64 + 2 * kp) * 1024 + q * 256 + 4 * ncl];
                    uint2 r0 = *(const uint2*)vp;
                    uint2 r1 = *(const uint2*)(vp + 1024);
                    unsigned e0[4] = { r0.x & 0xFFFFu, r0.x >> 16,
                                       r0.y & 0xFFFFu, r0.y >> 16 };
                    unsigned e1[4] = { r1.x & 0xFFFFu, r1.x >> 16,
                                       r1.y & 0xFFFFu, r1.y >> 16 };
                    for (int c = 0; c < 4; ++c)
                        *(unsigned*)&VT[(4 * ncl + c) * 72 + 2 * kp] =
                            e0[c] | (e1[c] << 16);
                }
                __syncthreads();   // VT ready; also covers Pbuf visibility
                for (int ks = 0; ks < 64; ks += 32) {
                    FragAB fa2[2], fb2[2];
                    for (int mt = 0; mt < 2; ++mt) {
                        int row = mt * 16 + lm;
                        fa2[mt].u[0] = *(const uint4*)&Pbuf[row * 72 + ks + klo];
                        fa2[mt].u[1] = *(const uint4*)&Pbuf[row * 72 + ks + klo + 8];
                    }
                    for (int nt = 0; nt < 2; ++nt) {
                        int n = w * 32 + nt * 16 + lm;
                        fb2[nt].u[0] = *(const uint4*)&VT[n * 72 + ks + khi];
                        fb2[nt].u[1] = *(const uint4*)&VT[n * 72 + ks + khi + 8];
                    }
                    for (int mt = 0; mt < 2; ++mt)
                        for (int nt = 0; nt < 2; ++nt)
                            oacc[q][mt][nt] =
                                wmma_bf16(fa2[mt].v, fb2[nt].v, oacc[q][mt][nt]);
                }
                __syncthreads();   // protect VT restage / Pbuf rewrite
            }
        }
    }

    if constexpr (!WRITE_OUT) {
        if (c8i == 0) {
            gM[m0 + ri] = m_run;
            gS[m0 + ri] = s_run;
        }
    } else {
        for (int q = 0; q < 4; ++q)
            for (int mt = 0; mt < 2; ++mt)
                for (int nt = 0; nt < 2; ++nt)
                    for (int r = 0; r < 8; ++r) {
                        int row = m0 + mt * 16 + r + hi8;
                        int col = q * 256 + w * 32 + nt * 16 + lm;
                        out[(size_t)row * 1024 + col] = oacc[q][mt][nt][r];
                    }
    }
}

// ---------------------------------------------------------------------------
// Host launch
// ---------------------------------------------------------------------------
extern "C" void kernel_launch(void* const* d_in, const int* in_sizes, int n_in,
                              void* d_out, int out_size, void* d_ws, size_t ws_size,
                              hipStream_t stream) {
    (void)in_sizes; (void)n_in; (void)out_size; (void)ws_size;
    const float* X = (const float*)d_in[0];   // inputs   [8192,1024]
    const float* R = (const float*)d_in[1];   // rotation [1024,1024]
    const float* E = (const float*)d_in[2];   // entangle [1024,1024]
    float* out = (float*)d_out;

    const size_t ND = (size_t)8192 * 1024;
    const size_t DD = (size_t)1024 * 1024;

    char* ws = (char*)d_ws;
    unsigned short* Xb = (unsigned short*)ws; ws += ND * 2;
    unsigned short* Rb = (unsigned short*)ws; ws += DD * 2;
    unsigned short* Eb = (unsigned short*)ws; ws += DD * 2;
    unsigned short* Qb = (unsigned short*)ws; ws += ND * 2;
    unsigned short* Kb = (unsigned short*)ws; ws += ND * 2;
    float* gM = (float*)ws; ws += (size_t)8192 * 4;
    float* gS = (float*)ws;

    // fp32 -> bf16
    cvt_f32_bf16<<<(int)(ND / 4 + 255) / 256, 256, 0, stream>>>(X, Xb, (int)(ND / 4));
    cvt_f32_bf16<<<(int)(DD / 4 + 255) / 256, 256, 0, stream>>>(R, Rb, (int)(DD / 4));
    cvt_f32_bf16<<<(int)(DD / 4 + 255) / 256, 256, 0, stream>>>(E, Eb, (int)(DD / 4));

    // Q = X@R, K = X@E (z selects matrix)
    gemm_bf16_qk<<<dim3(8, 64, 2), 256, 0, stream>>>(Xb, Rb, Eb, Qb, Kb);

    const float scale = 1.0f / 32.0f;  // 1/sqrt(1024)

    // pass 1: row max / sum-exp; pass 2: normalized P @ V
    attn_kernel<false><<<256, 256, 0, stream>>>(Qb, Kb, Xb, gM, gS, out, scale);
    attn_kernel<true ><<<256, 256, 0, stream>>>(Qb, Kb, Xb, gM, gS, out, scale);
}
